// MwAN_19851338842675
// MI455X (gfx1250) — matrix operand order repaired and checked by reading, not compiled, not run on hardware
//
#include <hip/hip_runtime.h>
#include <hip/hip_bf16.h>
#include <stdint.h>

// ---------------- CDNA5 WMMA types ----------------
typedef __bf16 bf16_t;
typedef bf16_t v16bf __attribute__((ext_vector_type(16)));
typedef float  v8f   __attribute__((ext_vector_type(8)));
typedef unsigned int u32x4 __attribute__((ext_vector_type(4)));

union Frag16 { u32x4 u[2]; v16bf v; };   // 32B = 16 bf16 per lane

__device__ __forceinline__ unsigned short f2bf(float f) {
  unsigned int x = __float_as_uint(f);
  x = (x + 0x7FFFu + ((x >> 16) & 1u)) >> 16;   // RNE
  return (unsigned short)x;
}
__device__ __forceinline__ float bf2f(unsigned short h) {
  return __uint_as_float(((unsigned int)h) << 16);
}
__device__ __forceinline__ float sigm(float x) { return 1.0f / (1.0f + __expf(-x)); }

// ---- CDNA5 async global->LDS mover (ASYNCcnt-tracked, 16B per lane) ----
__device__ __forceinline__ void async_b128(unsigned lds_off, const void* gaddr) {
  asm volatile("global_load_async_to_lds_b128 %0, %1, off"
               :: "v"(lds_off), "v"(gaddr) : "memory");
}
__device__ __forceinline__ void wait_async0() {
  asm volatile("s_wait_asynccnt 0x0" ::: "memory");
}
__device__ __forceinline__ unsigned lds_addr32(const void* p) {
  return (unsigned)(uintptr_t)p;   // low 32 bits of generic shared ptr = LDS offset
}

// =================================================================
// Elementwise helpers
// =================================================================
__global__ void k_cvt(const float* __restrict__ s, unsigned short* __restrict__ d, int n) {
  int i = blockIdx.x * 256 + threadIdx.x;
  if (i < n) d[i] = f2bf(s[i]);
}

__global__ void k_gather(const int* __restrict__ ids, const float* __restrict__ emb,
                         unsigned short* __restrict__ out, int n) {  // E = 256 fixed
  int i = blockIdx.x * 256 + threadIdx.x;
  if (i < n) {
    int t = i >> 8, e = i & 255;
    out[i] = f2bf(emb[(size_t)ids[t] * 256 + e]);
  }
}

__global__ void k_copy_slice(const float* __restrict__ src, float* __restrict__ dst,
                             unsigned short* __restrict__ dstbf, int cols, int ldd, int coff) {
  int i = blockIdx.x * 256 + threadIdx.x;
  int r = i / cols, c = i % cols;
  float v = src[(size_t)r * cols + c];
  size_t o = (size_t)r * ldd + coff + c;
  dst[o] = v;
  dstbf[o] = f2bf(v);
}

// =================================================================
// Shared-weight GEMM:  C(M,N) = act( A(M,K) @ W(N,K)^T + bias )
// A is bf16; mode 1/2 synthesize A on the fly:
//   m = b*(LqI*LpI) + q*LpI + p
//   mode 1 (prod): A[m][k] = X[b,p,k] * Y[b,q,k]
//   mode 2 (diff): A[m][k] = X[b,p,k] - Y[b,q,k]
// 128 threads = 4 waves; block tile 64x64; wave tile 16x64; K step 32.
// Tile staging uses GLOBAL_LOAD_ASYNC_TO_LDS_B128 (mode-0 A, and W always);
// tiles are pre-zeroed once so EXEC-masked edge rows stay zero.
// Requires K % 32 == 0 for mode 0 (holds for all call sites: 256 / 1536).
// =================================================================
__global__ void __launch_bounds__(128)
k_gemm(const unsigned short* __restrict__ A, const unsigned short* __restrict__ W,
       const float* __restrict__ bias, float* __restrict__ C, unsigned short* __restrict__ Cbf,
       int M, int N, int K, int ldc, int coff, int act, int mode,
       const unsigned short* __restrict__ X, const unsigned short* __restrict__ Y,
       int LqI, int LpI)
{
  __shared__ unsigned short At[64 * 40];
  __shared__ unsigned short Bt[64 * 40];
  int tid = threadIdx.x;
  int lane = tid & 31, wv = tid >> 5, hf = lane >> 4, l15 = lane & 15;
  int bm = blockIdx.y * 64, bn = blockIdx.x * 64;

  // zero once: async loads skip guarded (out-of-range) rows, which then stay 0
  for (int i = tid; i < 64 * 40; i += 128) { At[i] = 0; Bt[i] = 0; }
  __syncthreads();

  v8f acc[4];
  for (int i = 0; i < 4; i++) for (int j = 0; j < 8; j++) acc[i][j] = 0.0f;

  int ksteps = (K + 31) >> 5;
  for (int ks = 0; ks < ksteps; ks++) {
    int k0 = ks * 32;
    // ---- stage A tile (64 x 32 bf16) ----
    if (mode == 0) {
      for (int c = tid; c < 256; c += 128) {        // 256 async chunks of 16B
        int row = c >> 2, cc = c & 3;
        int gm = bm + row;
        if (gm < M)
          async_b128(lds_addr32(&At[row * 40 + cc * 8]),
                     A + (size_t)gm * K + k0 + cc * 8);
      }
    } else {
      for (int i = tid; i < 64 * 32; i += 128) {
        int row = i >> 5, kk = i & 31;
        int gm = bm + row, k = k0 + kk;
        unsigned short hv = 0;
        if (gm < M && k < K) {
          int bb = gm / (LqI * LpI), rem = gm % (LqI * LpI);
          int q = rem / LpI, p = rem % LpI;
          float xv = bf2f(X[((size_t)(bb * LpI + p)) * K + k]);
          float yv = bf2f(Y[((size_t)(bb * LqI + q)) * K + k]);
          hv = f2bf(mode == 1 ? xv * yv : xv - yv);
        }
        At[row * 40 + kk] = hv;
      }
    }
    // ---- stage W tile (64 n-rows x 32 k), always async ----
    for (int c = tid; c < 256; c += 128) {
      int row = c >> 2, cc = c & 3;
      int gn = bn + row;
      if (gn < N)
        async_b128(lds_addr32(&Bt[row * 40 + cc * 8]),
                   W + (size_t)gn * K + k0 + cc * 8);
    }
    wait_async0();
    __syncthreads();
    // ---- compute: A frag in contiguous k order (k = 16*hf + s, same as B), 4 N-tiles reuse it ----
    Frag16 a;
    int arow = wv * 16 + l15;
    a.u[0] = *(const u32x4*)(&At[arow * 40 + hf * 16]);
    a.u[1] = *(const u32x4*)(&At[arow * 40 + hf * 16 + 8]);
    for (int nt = 0; nt < 4; nt++) {
      Frag16 b;
      int brow = nt * 16 + l15;
      b.u[0] = *(const u32x4*)(&Bt[brow * 40 + hf * 16]);
      b.u[1] = *(const u32x4*)(&Bt[brow * 40 + hf * 16 + 8]);
      acc[nt] = __builtin_amdgcn_wmma_f32_16x16x32_bf16(
          false, a.v, false, b.v, (short)0, acc[nt], false, false);
    }
    __syncthreads();
  }
  // ---- epilogue ----
  for (int nt = 0; nt < 4; nt++) {
    int gn = bn + nt * 16 + l15;
    if (gn >= N) continue;
    float bv = bias ? bias[gn] : 0.0f;
    for (int v = 0; v < 8; v++) {
      int gm = bm + wv * 16 + v + 8 * hf;
      if (gm >= M) continue;
      float val = acc[nt][v] + bv;
      if (act == 1) val = (val > 0.0f) ? val : 0.01f * val;   // leaky_relu
      size_t idx = (size_t)gm * ldc + coff + gn;
      if (C)   C[idx] = val;
      if (Cbf) Cbf[idx] = f2bf(val);
    }
  }
}

// =================================================================
// Batched GEMM: C[b](M,N) = A[b](M,K) @ (transB ? B[b](N,K)^T : B[b](K,N))
// bf16 inputs, f32 (+optional bf16) output. Full K/M/N guards
// (K = 300/50/20 here, so scalar staging with per-element guards).
// =================================================================
__global__ void __launch_bounds__(128)
k_bgemm(const unsigned short* __restrict__ A, const unsigned short* __restrict__ B,
        float* __restrict__ C, unsigned short* __restrict__ Cbf,
        int M, int N, int K, int transB,
        long long sA, long long sB, long long sC, int ldc, int coff)
{
  __shared__ unsigned short At[64 * 40];
  __shared__ unsigned short Bt[64 * 40];
  int tid = threadIdx.x;
  int lane = tid & 31, wv = tid >> 5, hf = lane >> 4, l15 = lane & 15;
  int bm = blockIdx.y * 64, bn = blockIdx.x * 64;
  const unsigned short* Ab = A + (size_t)blockIdx.z * sA;
  const unsigned short* Bb = B + (size_t)blockIdx.z * sB;

  v8f acc[4];
  for (int i = 0; i < 4; i++) for (int j = 0; j < 8; j++) acc[i][j] = 0.0f;

  int ksteps = (K + 31) >> 5;
  for (int ks = 0; ks < ksteps; ks++) {
    int k0 = ks * 32;
    for (int i = tid; i < 64 * 32; i += 128) {
      int row = i >> 5, kk = i & 31;
      int gm = bm + row, k = k0 + kk;
      At[row * 40 + kk] = (gm < M && k < K) ? Ab[(size_t)gm * K + k] : (unsigned short)0;
    }
    for (int i = tid; i < 64 * 32; i += 128) {
      int row = i >> 5, kk = i & 31;
      int gn = bn + row, k = k0 + kk;
      unsigned short hv = 0;
      if (gn < N && k < K)
        hv = transB ? Bb[(size_t)gn * K + k] : Bb[(size_t)k * N + gn];
      Bt[row * 40 + kk] = hv;
    }
    __syncthreads();
    Frag16 a;
    int arow = wv * 16 + l15;
    a.u[0] = *(const u32x4*)(&At[arow * 40 + hf * 16]);
    a.u[1] = *(const u32x4*)(&At[arow * 40 + hf * 16 + 8]);
    for (int nt = 0; nt < 4; nt++) {
      Frag16 b;
      int brow = nt * 16 + l15;
      b.u[0] = *(const u32x4*)(&Bt[brow * 40 + hf * 16]);
      b.u[1] = *(const u32x4*)(&Bt[brow * 40 + hf * 16 + 8]);
      acc[nt] = __builtin_amdgcn_wmma_f32_16x16x32_bf16(
          false, a.v, false, b.v, (short)0, acc[nt], false, false);
    }
    __syncthreads();
  }
  for (int nt = 0; nt < 4; nt++) {
    int gn = bn + nt * 16 + l15;
    if (gn >= N) continue;
    for (int v = 0; v < 8; v++) {
      int gm = bm + wv * 16 + v + 8 * hf;
      if (gm >= M) continue;
      float val = acc[nt][v];
      size_t idx = (size_t)blockIdx.z * sC + (size_t)gm * ldc + coff + gn;
      if (C)   C[idx] = val;
      if (Cbf) Cbf[idx] = f2bf(val);
    }
  }
}

// =================================================================
// Fused GRU direction (torch gate math). H=128, 3H=384 fixed.
// 256 threads = 8 waves; wave wv owns hidden cols [16wv,16wv+16).
// Hidden state h lives in lane registers (the same lane that produces
// the WMMA D element (m,j) consumes it next step); bf16 shadow of h in
// LDS feeds the next step's A fragments. 3 WMMA (r,z,n) per m-tile/k-step.
// out[(m*T+tt)*ldo + coff + j] (+ bf16 copy), tt time-reversed if rev.
// =================================================================
__global__ void __launch_bounds__(256)
k_gru(const float* __restrict__ gi, const unsigned short* __restrict__ Wh,
      const float* __restrict__ bh, float* __restrict__ out,
      unsigned short* __restrict__ outbf,
      int Bsz, int T, int rev, int ldo, int coff)
{
  __shared__ unsigned short hbf[48 * 128];
  int tid = threadIdx.x, lane = tid & 31, wv = tid >> 5;
  int hf = lane >> 4, l15 = lane & 15;
  int MT = (Bsz + 15) >> 4;            // 1 (B=16) or 3 (B=48)
  int j = wv * 16 + l15;               // hidden column this lane owns

  float hreg[3][8];
  for (int mt = 0; mt < 3; mt++) for (int v = 0; v < 8; v++) hreg[mt][v] = 0.0f;
  for (int i = tid; i < 48 * 128; i += 256) hbf[i] = 0;
  __syncthreads();

  for (int t = 0; t < T; t++) {
    int tt = rev ? (T - 1 - t) : t;
    float hnew[3][8];
    for (int mt = 0; mt < MT; mt++) {
      v8f ar, az, an;
      for (int q = 0; q < 8; q++) { ar[q] = 0.0f; az[q] = 0.0f; an[q] = 0.0f; }
      for (int ks = 0; ks < 4; ks++) {             // K = 128
        Frag16 a;
        int arow = mt * 16 + l15;
        a.u[0] = *(const u32x4*)(&hbf[arow * 128 + ks * 32 + hf * 16]);
        a.u[1] = *(const u32x4*)(&hbf[arow * 128 + ks * 32 + hf * 16 + 8]);
        Frag16 br, bz, bn;
        const u32x4* pr = (const u32x4*)(Wh + (size_t)(j)       * 128 + ks * 32 + hf * 16);
        const u32x4* pz = (const u32x4*)(Wh + (size_t)(128 + j) * 128 + ks * 32 + hf * 16);
        const u32x4* pn = (const u32x4*)(Wh + (size_t)(256 + j) * 128 + ks * 32 + hf * 16);
        br.u[0] = pr[0]; br.u[1] = pr[1];
        bz.u[0] = pz[0]; bz.u[1] = pz[1];
        bn.u[0] = pn[0]; bn.u[1] = pn[1];
        ar = __builtin_amdgcn_wmma_f32_16x16x32_bf16(false, a.v, false, br.v, (short)0, ar, false, false);
        az = __builtin_amdgcn_wmma_f32_16x16x32_bf16(false, a.v, false, bz.v, (short)0, az, false, false);
        an = __builtin_amdgcn_wmma_f32_16x16x32_bf16(false, a.v, false, bn.v, (short)0, an, false, false);
      }
      for (int v = 0; v < 8; v++) {
        int m = mt * 16 + v + 8 * hf;
        float hn = 0.0f;
        if (m < Bsz) {
          size_t base = ((size_t)m * T + tt) * 384;
          float gir = gi[base + j], giz = gi[base + 128 + j], gin = gi[base + 256 + j];
          float ghr = ar[v] + bh[j], ghz = az[v] + bh[128 + j], ghn = an[v] + bh[256 + j];
          float r = sigm(gir + ghr);
          float z = sigm(giz + ghz);
          float n = tanhf(gin + r * ghn);
          hn = (1.0f - z) * n + z * hreg[mt][v];
          hreg[mt][v] = hn;
          size_t o = ((size_t)m * T + tt) * ldo + coff + j;
          out[o] = hn;
          if (outbf) outbf[o] = f2bf(hn);
        }
        hnew[mt][v] = hn;
      }
    }
    __syncthreads();                     // all reads of hbf done
    for (int mt = 0; mt < MT; mt++)
      for (int v = 0; v < 8; v++) {
        int m = mt * 16 + v + 8 * hf;
        if (m < Bsz) hbf[(size_t)m * 128 + j] = f2bf(hnew[mt][v]);
      }
    __syncthreads();                     // new h visible
  }
}

// =================================================================
// Row softmax (in place) + bf16 copy. grid.x = rows.
// =================================================================
__global__ void __launch_bounds__(256)
k_softmax(float* __restrict__ x, unsigned short* __restrict__ xbf, int N)
{
  __shared__ float red[256];
  int row = blockIdx.x, tid = threadIdx.x;
  float* xr = x + (size_t)row * N;
  float mx = -1e30f;
  for (int i = tid; i < N; i += 256) mx = fmaxf(mx, xr[i]);
  red[tid] = mx; __syncthreads();
  for (int s = 128; s > 0; s >>= 1) { if (tid < s) red[tid] = fmaxf(red[tid], red[tid + s]); __syncthreads(); }
  mx = red[0]; __syncthreads();
  float sum = 0.0f;
  for (int i = tid; i < N; i += 256) { float e = __expf(xr[i] - mx); xr[i] = e; sum += e; }
  red[tid] = sum; __syncthreads();
  for (int s = 128; s > 0; s >>= 1) { if (tid < s) red[tid] += red[tid + s]; __syncthreads(); }
  float inv = 1.0f / red[0];
  for (int i = tid; i < N; i += 256) {
    float p = xr[i] * inv;
    xr[i] = p;
    if (xbf) xbf[(size_t)row * N + i] = f2bf(p);
  }
}

// out[r] = sum_j f(t1[r,j]) * v[j],  f = tanh or identity.  grid.x = rows.
__global__ void __launch_bounds__(256)
k_rowdot(const float* __restrict__ t1, const float* __restrict__ v,
         float* __restrict__ out, int Hn, int use_tanh)
{
  __shared__ float red[256];
  int r = blockIdx.x, tid = threadIdx.x;
  float s = 0.0f;
  for (int jj = tid; jj < Hn; jj += 256) {
    float x = t1[(size_t)r * Hn + jj];
    if (use_tanh) x = tanhf(x);
    s += x * v[jj];
  }
  red[tid] = s; __syncthreads();
  for (int st = 128; st > 0; st >>= 1) { if (tid < st) red[tid] += red[tid + st]; __syncthreads(); }
  if (tid == 0) out[r] = red[0];
}

// concat attention scores: out[b,q,p] = sum_j tanh(s1[b,p,j]+s2[b,q,j])*vc[j], H=128
__global__ void __launch_bounds__(128)
k_concat_score(const float* __restrict__ s1, const float* __restrict__ s2,
               const float* __restrict__ vc, float* __restrict__ out, int Lq, int Lp)
{
  __shared__ float red[128];
  int r = blockIdx.x;
  int p = r % Lp, bq = r / Lp, q = bq % Lq, b = bq / Lq;
  int tid = threadIdx.x;
  float x = tanhf(s1[((size_t)(b * Lp + p)) * 128 + tid] +
                  s2[((size_t)(b * Lq + q)) * 128 + tid]) * vc[tid];
  red[tid] = x; __syncthreads();
  for (int st = 64; st > 0; st >>= 1) { if (tid < st) red[tid] += red[tid + st]; __syncthreads(); }
  if (tid == 0) out[r] = red[0];
}

// prediction scores: out[b*Lq+q] = sum_j (p1[b,q,j] + p2[b,j]) * vp[j], H=128
__global__ void __launch_bounds__(128)
k_pred_score(const float* __restrict__ p1, const float* __restrict__ p2,
             const float* __restrict__ vp, float* __restrict__ out, int Lq)
{
  __shared__ float red[128];
  int r = blockIdx.x, b = r / Lq, tid = threadIdx.x;
  float x = (p1[(size_t)r * 128 + tid] + p2[(size_t)b * 128 + tid]) * vp[tid];
  red[tid] = x; __syncthreads();
  for (int st = 64; st > 0; st >>= 1) { if (tid < st) red[tid] += red[tid + st]; __syncthreads(); }
  if (tid == 0) out[r] = red[0];
}

// final: logits[b,k] = a_rep[b*3+k] . enc[b]; softmax over k; loss = mean(-log p0)
__global__ void k_final(const float* __restrict__ a_rep, const float* __restrict__ enc,
                        float* __restrict__ out, int B)
{
  __shared__ float logits[48];
  __shared__ float losses[16];
  int tid = threadIdx.x;
  if (tid < 3 * B) {
    int b = tid / 3, k = tid % 3;
    float s = 0.0f;
    for (int d = 0; d < 256; d++)
      s += a_rep[(size_t)(b * 3 + k) * 256 + d] * enc[(size_t)b * 256 + d];
    logits[tid] = s;
  }
  __syncthreads();
  if (tid < B) {
    float l0 = logits[tid * 3], l1 = logits[tid * 3 + 1], l2 = logits[tid * 3 + 2];
    float m = fmaxf(l0, fmaxf(l1, l2));
    float e0 = __expf(l0 - m), e1 = __expf(l1 - m), e2 = __expf(l2 - m);
    losses[tid] = -logf(e0 / (e0 + e1 + e2));
  }
  __syncthreads();
  if (tid == 0) { float s = 0.0f; for (int b = 0; b < B; b++) s += losses[b]; out[0] = s / B; }
}

// =================================================================
// Orchestration
// =================================================================
extern "C" void kernel_launch(void* const* d_in, const int* in_sizes, int n_in,
                              void* d_out, int out_size, void* d_ws, size_t ws_size,
                              hipStream_t stream) {
  (void)in_sizes; (void)n_in; (void)out_size; (void)ws_size;
  const int B = 16, Lq = 50, Lp = 300, La = 20, E = 256, H = 128, H2 = 256,
            H3 = 384, H12 = 1536, G = 48;

  const int*   query   = (const int*)d_in[0];
  const int*   passage = (const int*)d_in[1];
  const int*   answer  = (const int*)d_in[2];
  const float* emb     = (const float*)d_in[4];
  // q_enc / p_enc / a_enc / agg : (Wif, Whf, bif, bhf, Wib, Whb, bib, bhb)
  const float* qW[8]; const float* pW[8]; const float* aW[8]; const float* gW[8];
  for (int i = 0; i < 8; i++) { qW[i] = (const float*)d_in[5 + i];  pW[i] = (const float*)d_in[13 + i];
                                aW[i] = (const float*)d_in[21 + i]; gW[i] = (const float*)d_in[29 + i]; }
  const float* a_att = (const float*)d_in[37];
  const float* Wc1 = (const float*)d_in[38]; const float* Wc2 = (const float*)d_in[39];
  const float* vc  = (const float*)d_in[40]; const float* Wb  = (const float*)d_in[41];
  const float* Wd  = (const float*)d_in[42]; const float* vd  = (const float*)d_in[43];
  const float* Wm  = (const float*)d_in[44]; const float* vm  = (const float*)d_in[45];
  const float* Ws  = (const float*)d_in[46]; const float* vs  = (const float*)d_in[47];
  const float* Wq  = (const float*)d_in[48]; const float* vq  = (const float*)d_in[49];
  const float* Wp1 = (const float*)d_in[50]; const float* Wp2 = (const float*)d_in[51];
  const float* vp  = (const float*)d_in[52]; const float* Wpred = (const float*)d_in[53];

  // ---- bump allocator over d_ws (~185 MB peak) ----
  char* ws = (char*)d_ws; size_t off = 0;
  auto alloc = [&](size_t bytes) -> void* {
    void* p = ws + off; off = (off + bytes + 255) & ~(size_t)255; return p;
  };
  auto AF = [&](size_t n) { return (float*)alloc(n * 4); };
  auto AB = [&](size_t n) { return (unsigned short*)alloc(n * 2); };

  auto cvt = [&](const float* s, unsigned short* d, size_t n) {
    k_cvt<<<(unsigned)((n + 255) / 256), 256, 0, stream>>>(s, d, (int)n);
  };
  auto gemm = [&](const unsigned short* A, const unsigned short* W, const float* bias,
                  float* C, unsigned short* Cbf, int M, int N, int K, int ldc, int coff,
                  int act, int mode, const unsigned short* X, const unsigned short* Y,
                  int LqI, int LpI) {
    dim3 g((N + 63) / 64, (M + 63) / 64);
    k_gemm<<<g, 128, 0, stream>>>(A, W, bias, C, Cbf, M, N, K, ldc, coff, act, mode, X, Y, LqI, LpI);
  };
  auto bgemm = [&](const unsigned short* A, const unsigned short* Bm, float* C,
                   unsigned short* Cbf, int M, int N, int K, int transB,
                   long long sA, long long sB, long long sC, int ldc, int coff, int batches) {
    dim3 g((N + 63) / 64, (M + 63) / 64, batches);
    k_bgemm<<<g, 128, 0, stream>>>(A, Bm, C, Cbf, M, N, K, transB, sA, sB, sC, ldc, coff);
  };

  // ---- bf16 weights ----
  unsigned short *wqif = AB(98304), *wqhf = AB(49152), *wqib = AB(98304), *wqhb = AB(49152);
  unsigned short *wpif = AB(98304), *wphf = AB(49152), *wpib = AB(98304), *wphb = AB(49152);
  unsigned short *waif = AB(98304), *wahf = AB(49152), *waib = AB(98304), *wahb = AB(49152);
  unsigned short *wgif = AB(589824), *wghf = AB(49152), *wgib = AB(589824), *wghb = AB(49152);
  unsigned short *bWc1 = AB(32768), *bWc2 = AB(32768), *bWb = AB(65536), *bWd = AB(32768);
  unsigned short *bWm = AB(32768), *bWs = AB(32768), *bWq = AB(32768);
  unsigned short *bWp1 = AB(32768), *bWp2 = AB(32768), *bpred = AB(65536);
  cvt(qW[0], wqif, 98304); cvt(qW[1], wqhf, 49152); cvt(qW[4], wqib, 98304); cvt(qW[5], wqhb, 49152);
  cvt(pW[0], wpif, 98304); cvt(pW[1], wphf, 49152); cvt(pW[4], wpib, 98304); cvt(pW[5], wphb, 49152);
  cvt(aW[0], waif, 98304); cvt(aW[1], wahf, 49152); cvt(aW[4], waib, 98304); cvt(aW[5], wahb, 49152);
  cvt(gW[0], wgif, 589824); cvt(gW[1], wghf, 49152); cvt(gW[4], wgib, 589824); cvt(gW[5], wghb, 49152);
  cvt(Wc1, bWc1, 32768); cvt(Wc2, bWc2, 32768); cvt(Wb, bWb, 65536); cvt(Wd, bWd, 32768);
  cvt(Wm, bWm, 32768); cvt(Ws, bWs, 32768); cvt(Wq, bWq, 32768);
  cvt(Wp1, bWp1, 32768); cvt(Wp2, bWp2, 32768); cvt(Wpred, bpred, 65536);

  // ---- embedding gathers (note reference swap: passage->q_enc, query->p_enc) ----
  unsigned short *qe = AB((size_t)B * Lq * E), *pe = AB((size_t)B * Lp * E), *ae = AB((size_t)G * La * E);
  k_gather<<<(B * Lq * E) / 256, 256, 0, stream>>>(query, emb, qe, B * Lq * E);
  k_gather<<<(B * Lp * E) / 256, 256, 0, stream>>>(passage, emb, pe, B * Lp * E);
  k_gather<<<(G * La * E) / 256, 256, 0, stream>>>(answer, emb, ae, G * La * E);

  // ---- input-projection (gi) GEMMs ----
  float *gi_hq_f = AF((size_t)B * Lp * H3), *gi_hq_b = AF((size_t)B * Lp * H3);
  float *gi_hp_f = AF((size_t)B * Lq * H3), *gi_hp_b = AF((size_t)B * Lq * H3);
  float *gi_a_f  = AF((size_t)G * La * H3), *gi_a_b  = AF((size_t)G * La * H3);
  gemm(pe, wqif, qW[2], gi_hq_f, nullptr, B * Lp, H3, E, H3, 0, 0, 0, nullptr, nullptr, 0, 0);
  gemm(pe, wqib, qW[6], gi_hq_b, nullptr, B * Lp, H3, E, H3, 0, 0, 0, nullptr, nullptr, 0, 0);
  gemm(qe, wpif, pW[2], gi_hp_f, nullptr, B * Lq, H3, E, H3, 0, 0, 0, nullptr, nullptr, 0, 0);
  gemm(qe, wpib, pW[6], gi_hp_b, nullptr, B * Lq, H3, E, H3, 0, 0, 0, nullptr, nullptr, 0, 0);
  gemm(ae, waif, aW[2], gi_a_f, nullptr, G * La, H3, E, H3, 0, 0, 0, nullptr, nullptr, 0, 0);
  gemm(ae, waib, aW[6], gi_a_b, nullptr, G * La, H3, E, H3, 0, 0, 0, nullptr, nullptr, 0, 0);

  // ---- bi-GRUs (fused recurrent WMMA kernels) ----
  float *hq = AF((size_t)B * Lp * H2);  unsigned short *hq_bf = AB((size_t)B * Lp * H2);
  float *hp = AF((size_t)B * Lq * H2);  unsigned short *hp_bf = AB((size_t)B * Lq * H2);
  float *a_h = AF((size_t)G * La * H2); unsigned short *a_h_bf = AB((size_t)G * La * H2);
  k_gru<<<1, 256, 0, stream>>>(gi_hq_f, wqhf, qW[3], hq, hq_bf, B, Lp, 0, H2, 0);
  k_gru<<<1, 256, 0, stream>>>(gi_hq_b, wqhb, qW[7], hq, hq_bf, B, Lp, 1, H2, H);
  k_gru<<<1, 256, 0, stream>>>(gi_hp_f, wphf, pW[3], hp, hp_bf, B, Lq, 0, H2, 0);
  k_gru<<<1, 256, 0, stream>>>(gi_hp_b, wphb, pW[7], hp, hp_bf, B, Lq, 1, H2, H);
  k_gru<<<1, 256, 0, stream>>>(gi_a_f, wahf, aW[3], a_h, a_h_bf, G, La, 0, H2, 0);
  k_gru<<<1, 256, 0, stream>>>(gi_a_b, wahb, aW[7], a_h, a_h_bf, G, La, 1, H2, H);

  // ---- shared projections of hq / hp ----
  float *s1 = AF((size_t)B * Lp * H), *s2 = AF((size_t)B * Lq * H), *hqWq = AF((size_t)B * Lp * H);
  unsigned short *hqWb_bf = AB((size_t)B * Lp * H2);
  gemm(hq_bf, bWc1, nullptr, s1, nullptr, B * Lp, H, H2, H, 0, 0, 0, nullptr, nullptr, 0, 0);
  gemm(hp_bf, bWc2, nullptr, s2, nullptr, B * Lq, H, H2, H, 0, 0, 0, nullptr, nullptr, 0, 0);
  gemm(hq_bf, bWb, nullptr, nullptr, hqWb_bf, B * Lp, H2, H2, H2, 0, 0, 0, nullptr, nullptr, 0, 0);
  gemm(hq_bf, bWq, nullptr, hqWq, nullptr, B * Lp, H, H2, H, 0, 0, 0, nullptr, nullptr, 0, 0);

  // ---- attention scores ----
  float *sjt_c = AF((size_t)B * Lq * Lp), *sjt_b = AF((size_t)B * Lq * Lp);
  float *sjt_d = AF((size_t)B * Lq * Lp), *sjt_m = AF((size_t)B * Lq * Lp);
  float *sjt_s = AF((size_t)B * Lq * Lq);
  unsigned short *pc = AB((size_t)B * Lq * Lp), *pb = AB((size_t)B * Lq * Lp);
  unsigned short *pd = AB((size_t)B * Lq * Lp), *pm = AB((size_t)B * Lq * Lp);
  unsigned short *ps = AB((size_t)B * Lq * Lq);
  float *t1 = AF((size_t)B * Lq * Lp * H);   // reused: dot -> minus -> self

  k_concat_score<<<B * Lq * Lp, 128, 0, stream>>>(s1, s2, vc, sjt_c, Lq, Lp);
  k_softmax<<<B * Lq, 256, 0, stream>>>(sjt_c, pc, Lp);

  bgemm(hp_bf, hqWb_bf, sjt_b, nullptr, Lq, Lp, H2, 1,
        (long long)Lq * H2, (long long)Lp * H2, (long long)Lq * Lp, Lp, 0, B);
  k_softmax<<<B * Lq, 256, 0, stream>>>(sjt_b, pb, Lp);

  gemm(nullptr, bWd, nullptr, t1, nullptr, B * Lq * Lp, H, H2, H, 0, 0, 1, hq_bf, hp_bf, Lq, Lp);
  k_rowdot<<<B * Lq * Lp, 256, 0, stream>>>(t1, vd, sjt_d, H, 1);
  k_softmax<<<B * Lq, 256, 0, stream>>>(sjt_d, pd, Lp);

  gemm(nullptr, bWm, nullptr, t1, nullptr, B * Lq * Lp, H, H2, H, 0, 0, 2, hq_bf, hp_bf, Lq, Lp);
  k_rowdot<<<B * Lq * Lp, 256, 0, stream>>>(t1, vm, sjt_m, H, 1);
  k_softmax<<<B * Lq, 256, 0, stream>>>(sjt_m, pm, Lp);

  gemm(nullptr, bWs, nullptr, t1, nullptr, B * Lq * Lq, H, H2, H, 0, 0, 1, hp_bf, hp_bf, Lq, Lq);
  k_rowdot<<<B * Lq * Lq, 256, 0, stream>>>(t1, vs, sjt_s, H, 1);
  k_softmax<<<B * Lq, 256, 0, stream>>>(sjt_s, ps, Lq);

  // ---- aggregation = [hp, qts, qtc, qtd, qtb, qtm] (B,Lq,12H) ----
  float *aggf = AF((size_t)B * Lq * H12); unsigned short *aggbf = AB((size_t)B * Lq * H12);
  k_copy_slice<<<(B * Lq * H2) / 256, 256, 0, stream>>>(hp, aggf, aggbf, H2, H12, 0);
  long long sQ = (long long)Lq * H12;
  bgemm(ps, hp_bf, aggf, aggbf, Lq, H2, Lq, 0, (long long)Lq * Lq, (long long)Lq * H2, sQ, H12, 256, B);
  bgemm(pc, hq_bf, aggf, aggbf, Lq, H2, Lp, 0, (long long)Lq * Lp, (long long)Lp * H2, sQ, H12, 512, B);
  bgemm(pd, hq_bf, aggf, aggbf, Lq, H2, Lp, 0, (long long)Lq * Lp, (long long)Lp * H2, sQ, H12, 768, B);
  bgemm(pb, hq_bf, aggf, aggbf, Lq, H2, Lp, 0, (long long)Lq * Lp, (long long)Lp * H2, sQ, H12, 1024, B);
  bgemm(pm, hq_bf, aggf, aggbf, Lq, H2, Lp, 0, (long long)Lq * Lp, (long long)Lp * H2, sQ, H12, 1280, B);

  // ---- aggregation bi-GRU ----
  float *gi_ag_f = AF((size_t)B * Lq * H3), *gi_ag_b = AF((size_t)B * Lq * H3);
  float *agg_rep = AF((size_t)B * Lq * H2); unsigned short *agg_rep_bf = AB((size_t)B * Lq * H2);
  gemm(aggbf, wgif, gW[2], gi_ag_f, nullptr, B * Lq, H3, H12, H3, 0, 0, 0, nullptr, nullptr, 0, 0);
  gemm(aggbf, wgib, gW[6], gi_ag_b, nullptr, B * Lq, H3, H12, H3, 0, 0, 0, nullptr, nullptr, 0, 0);
  k_gru<<<1, 256, 0, stream>>>(gi_ag_f, wghf, gW[3], agg_rep, agg_rep_bf, B, Lq, 0, H2, 0);
  k_gru<<<1, 256, 0, stream>>>(gi_ag_b, wghb, gW[7], agg_rep, agg_rep_bf, B, Lq, 1, H2, H);

  // ---- prediction head ----
  float *spred = AF((size_t)B * Lp); unsigned short *spred_bf = AB((size_t)B * Lp);
  float *rq = AF((size_t)B * H2); unsigned short *rq_bf = AB((size_t)B * H2);
  float *p1 = AF((size_t)B * Lq * H), *p2 = AF((size_t)B * H);
  float *tvec = AF((size_t)B * Lq); unsigned short *tvec_bf = AB((size_t)B * Lq);
  float *rp = AF((size_t)B * H2); unsigned short *rp_bf = AB((size_t)B * H2);
  float *enc = AF((size_t)B * H2);

  k_rowdot<<<B * Lp, 256, 0, stream>>>(hqWq, vq, spred, H, 1);
  k_softmax<<<B, 256, 0, stream>>>(spred, spred_bf, Lp);
  bgemm(spred_bf, hq_bf, rq, rq_bf, 1, H2, Lp, 0, (long long)Lp, (long long)Lp * H2, H2, H2, 0, B);
  gemm(agg_rep_bf, bWp1, nullptr, p1, nullptr, B * Lq, H, H2, H, 0, 0, 0, nullptr, nullptr, 0, 0);
  gemm(rq_bf, bWp2, nullptr, p2, nullptr, B, H, H2, H, 0, 0, 0, nullptr, nullptr, 0, 0);
  k_pred_score<<<B * Lq, 128, 0, stream>>>(p1, p2, vp, tvec, Lq);
  k_softmax<<<B, 256, 0, stream>>>(tvec, tvec_bf, Lq);
  bgemm(tvec_bf, agg_rep_bf, rp, rp_bf, 1, H2, Lq, 0, (long long)Lq, (long long)Lq * H2, H2, H2, 0, B);
  gemm(rp_bf, bpred, nullptr, enc, nullptr, B, E, H2, E, 0, 1 /*leaky*/, 0, nullptr, nullptr, 0, 0);

  // ---- answer pooling + loss ----
  float *a_sc = AF((size_t)G * La); unsigned short *a_sc_bf = AB((size_t)G * La);
  float *a_rep = AF((size_t)G * H2);
  k_rowdot<<<G * La, 256, 0, stream>>>(a_h, a_att, a_sc, H2, 0);
  k_softmax<<<G, 256, 0, stream>>>(a_sc, a_sc_bf, La);
  bgemm(a_sc_bf, a_h_bf, a_rep, nullptr, 1, H2, La, 0, (long long)La, (long long)La * H2, H2, H2, 0, G);

  k_final<<<1, 64, 0, stream>>>(a_rep, enc, (float*)d_out, B);
}
